// DigitConvolutionalModel_2156073582717
// MI455X (gfx1250) — compile-verified, matrix operand order checked by reading, not run on hardware
//
#include <hip/hip_runtime.h>

// ---------------------------------------------------------------------------
// DigitConvolutionalModel on MI455X (gfx1250, wave32, WMMA 16x16x32 f16)
//
// Conv is linear -> folded into layer 1: W1eff = C_conv @ W1 (784x100).
// out = relu(relu(x@W1eff+b1)@W2+b2)@W3+b3.
// Pad: K1 784->800, N1/K2 100->112/128, N3 10->16.
//
// v4: all LDS traffic via direct __shared__ array indexing (v3's pointer
// array lost addrspace(3) and the slab path lowered to global ops).
// Double-buffered 8KB W1eff K-slabs: global_load_b128 issued before the
// WMMAs of the current slab, ds_store_b128 after, barrier per slab.
// W2T/W3T LDS-resident; A register-double-buffered NT loads; NT out stores.
// ---------------------------------------------------------------------------

typedef __attribute__((ext_vector_type(16))) _Float16 v16h;
typedef __attribute__((ext_vector_type(8)))  float    v8f;
typedef __attribute__((ext_vector_type(4)))  unsigned int v4u;

#define BATCH 32768
#define K1    800       // padded 784
#define N1P   112       // padded 100
#define K2    128       // padded 100
#define NW    8         // waves per block in fused kernel
#define NSLAB 25        // K1/32
#define SLAB_E     4096 // elements per slab buffer (8192 B, 1KB tail pad)
#define SLAB_BYTES (SLAB_E * 2)
#define HB_E       (16 * K2)  // per-wave h tile elements

// ---- ws layout (bytes) ----
#define WS_XH   0                    // [32768][800] f16 = 52,428,800 B
#define WS_W1S  52428800             // [25][4096]  f16 =    204,800 B (padded slabs)
#define WS_W2T  52633600             // [112][128]  f16 =     28,672 B
#define WS_W3T  52662272             // [16][128]   f16 =      4,096 B

// Load a 16x32 f16 fragment from a named __shared__ array at element index e
// (direct indexing so addrspace(3) survives inlining -> ds_load_b128 x2).
#define LDS_FRAG(arr, e)                                   \
  ({ union { v4u q[2]; v16h v; } u_;                       \
     u_.q[0] = *(const v4u*)&arr[(e)];                     \
     u_.q[1] = *(const v4u*)&arr[(e) + 16];                \
     u_.v; })

// 16x32 f16 fragment from global (ISA 16-bit A layout: lanes 0-15 hold
// K {0..7,16..23}, lanes 16-31 hold K {8..15,24..31}; lane&15 = row).
__device__ __forceinline__ v16h load_frag_nt(const _Float16* __restrict base,
                                             int ldk, int row, int khalf, int k0) {
  union { v4u q[2]; v16h v; } u;
  const _Float16* p = base + (size_t)row * ldk + k0 + khalf * 8;
  u.q[0] = __builtin_nontemporal_load((const v4u*)(p));
  u.q[1] = __builtin_nontemporal_load((const v4u*)(p + 16));
  return u.v;
}

// ---------------------------------------------------------------------------
// Kernel 1: f16 transposed+padded weights; W1eff folds the 3x3 conv and is
// stored as padded contiguous K-slabs: w1s[(k/32)*4096 + n*32 + k%32].
// ---------------------------------------------------------------------------
__global__ void prep_weights(const float* __restrict wconv,
                             const float* __restrict W1,
                             const float* __restrict W2,
                             const float* __restrict W3,
                             _Float16* __restrict w1s,
                             _Float16* __restrict w2t,
                             _Float16* __restrict w3t) {
  int idx = blockIdx.x * blockDim.x + threadIdx.x;
  if (idx < N1P * K1) {                       // W1effT value at (n, k)
    int n = idx / K1, k = idx % K1;
    float acc = 0.f;
    if (n < 100 && k < 784) {
      int py = k / 28, px = k % 28;
      #pragma unroll
      for (int dy = 0; dy < 3; ++dy)
        #pragma unroll
        for (int dx = 0; dx < 3; ++dx) {
          int oy = py - dy, ox = px - dx;
          if (oy >= 0 && oy < 26 && ox >= 0 && ox < 26)
            acc += wconv[dy * 3 + dx] * W1[(oy * 26 + ox) * 100 + n];
        }
    }
    w1s[(k >> 5) * SLAB_E + n * 32 + (k & 31)] = (_Float16)acc;
  } else if (idx < N1P * K1 + N1P * K2) {     // W2T[n][k]
    int j = idx - N1P * K1;
    int n = j / K2, k = j % K2;
    w2t[j] = (_Float16)((n < 100 && k < 100) ? W2[k * 100 + n] : 0.f);
  } else if (idx < N1P * K1 + N1P * K2 + 16 * K2) {  // W3T[n][k]
    int j = idx - (N1P * K1 + N1P * K2);
    int n = j / K2, k = j % K2;
    w3t[j] = (_Float16)((n < 10 && k < 100) ? W3[k * 10 + n] : 0.f);
  }
}

// ---------------------------------------------------------------------------
// Kernel 2: cast x f32 -> f16, pad row 784 -> 800 with zeros. 8 elems/thread.
// ---------------------------------------------------------------------------
__global__ void cast_x(const float* __restrict x, _Float16* __restrict xh) {
  long long t = (long long)blockIdx.x * blockDim.x + threadIdx.x; // < 32768*100
  int b = (int)(t / 100);
  int k = (int)(t % 100) * 8;
  const float* src = x + (long long)b * 784 + k;
  _Float16 tmp[8];
  #pragma unroll
  for (int i = 0; i < 8; ++i) {
    int kk = k + i;
    tmp[i] = (kk < 784) ? (_Float16)src[i] : (_Float16)0.f;
  }
  *(v4u*)(xh + (long long)b * K1 + k) = *(const v4u*)tmp;
}

// ---------------------------------------------------------------------------
// Kernel 3: fused 3-layer MLP. One wave owns a 16-row batch strip.
// GEMM1: 25x7 wmma (B from double-buffered LDS slabs);
// GEMM2: 28 wmma; GEMM3: 4 wmma (B from LDS-resident W2T/W3T).
// ---------------------------------------------------------------------------
__global__ __launch_bounds__(256) void mlp_fused(
    const _Float16* __restrict xh,   // [B][800]
    const _Float16* __restrict w1s,  // [25][4096] padded slabs
    const _Float16* __restrict w2t,  // [112][128]
    const _Float16* __restrict w3t,  // [16][128]
    const float* __restrict b1, const float* __restrict b2,
    const float* __restrict b3, float* __restrict out) {

  __shared__ __align__(16) _Float16 w1slab[2 * SLAB_E];    // 16 KB
  __shared__ __align__(16) _Float16 w2s[N1P * K2];         // 28 KB
  __shared__ __align__(16) _Float16 w3s[16 * K2];          // 4 KB
  __shared__ __align__(16) _Float16 hbuf[NW * HB_E];       // 32 KB

  const int tid   = threadIdx.x;
  const int lane  = tid & 31;
  const int wave  = tid >> 5;
  const int mrow  = lane & 15;
  const int khalf = (lane >> 4) & 1;
  const int m0    = (blockIdx.x * NW + wave) * 16;
  const int hb    = wave * HB_E;

  // ---- stage W2T, W3T and W1 slab 0 into LDS (one-time) ----
  for (int i = tid; i < (N1P * K2) / 8; i += 256)          // 1792 chunks
    *(v4u*)&w2s[i * 8] = *(const v4u*)((const char*)w2t + (size_t)i * 16);
  for (int i = tid; i < (16 * K2) / 8; i += 256)           // 256 chunks
    *(v4u*)&w3s[i * 8] = *(const v4u*)((const char*)w3t + (size_t)i * 16);
  {
    #pragma unroll
    for (int j = 0; j < 2; ++j) {                          // 512 chunks
      int i = tid + j * 256;
      *(v4u*)&w1slab[i * 8] = *(const v4u*)((const char*)w1s + (size_t)i * 16);
    }
  }

  // Zero per-wave staging tile so padded K columns (112..127) read as 0.
  {
    v4u z = {0u, 0u, 0u, 0u};
    #pragma unroll
    for (int i = 0; i < 8; ++i)
      *(v4u*)&hbuf[hb + (i * 32 + lane) * 8] = z;
  }

  float b1v[7], b2v[7];
  #pragma unroll
  for (int n = 0; n < 7; ++n) {
    int col = n * 16 + mrow;
    b1v[n] = (col < 100) ? b1[col] : 0.f;
    b2v[n] = (col < 100) ? b2[col] : 0.f;
  }
  float b3v = (mrow < 10) ? b3[mrow] : 0.f;

  const v8f zacc = {0.f, 0.f, 0.f, 0.f, 0.f, 0.f, 0.f, 0.f};

  __syncthreads();

  // ---------------- layer 1: [16x800] x [800x112] ----------------
  v8f acc[7];
  #pragma unroll
  for (int n = 0; n < 7; ++n) acc[n] = zacc;

  const _Float16* arow = xh + (size_t)m0 * K1;

  int cur = 0;
  v16h a = load_frag_nt(arow, K1, mrow, khalf, 0);
  for (int ks = 0; ks < NSLAB; ++ks) {
    const bool have_next = (ks + 1 < NSLAB);
    // Issue global loads for the next slab BEFORE the WMMAs so L2 latency
    // hides behind matrix work; ds_store + barrier afterwards.
    v4u s0 = {0u, 0u, 0u, 0u}, s1 = {0u, 0u, 0u, 0u};
    if (have_next) {
      const char* g = (const char*)w1s + (size_t)(ks + 1) * SLAB_BYTES;
      s0 = *(const v4u*)(g + (size_t)tid * 16);
      s1 = *(const v4u*)(g + (size_t)(tid + 256) * 16);
      __builtin_prefetch(arow + (size_t)mrow * K1 + (ks + 1) * 32, 0, 1);
    }
    v16h an = a;
    if (have_next) an = load_frag_nt(arow, K1, mrow, khalf, (ks + 1) * 32);

    const int sb = cur * SLAB_E;
    #pragma unroll
    for (int n = 0; n < 7; ++n) {
      v16h b = LDS_FRAG(w1slab, sb + (n * 16 + mrow) * 32 + khalf * 8);
      acc[n] = __builtin_amdgcn_wmma_f32_16x16x32_f16(
          false, a, false, b, (short)0, acc[n], false, false);
    }
    if (have_next) {
      const int db = (cur ^ 1) * SLAB_E;
      *(v4u*)&w1slab[db + tid * 8] = s0;
      *(v4u*)&w1slab[db + (tid + 256) * 8] = s1;
    }
    __syncthreads();
    a = an;
    cur ^= 1;
  }

  // bias + ReLU, relayout C (col-per-lane) -> row-major f16 tile in LDS
  #pragma unroll
  for (int n = 0; n < 7; ++n) {
    int N = n * 16 + mrow;
    #pragma unroll
    for (int r = 0; r < 8; ++r) {
      int M = khalf * 8 + r;
      float v = acc[n][r] + b1v[n];
      hbuf[hb + M * K2 + N] = (_Float16)(v > 0.f ? v : 0.f);
    }
  }
  asm volatile("s_wait_dscnt 0x0" ::: "memory");  // h1 visible within wave

  // ---------------- layer 2: [16x128] x [128x112] ----------------
  v8f acc2[7];
  #pragma unroll
  for (int n = 0; n < 7; ++n) acc2[n] = zacc;

  for (int k = 0; k < K2; k += 32) {
    v16h a2 = LDS_FRAG(hbuf, hb + mrow * K2 + k + khalf * 8);
    #pragma unroll
    for (int n = 0; n < 7; ++n) {
      v16h b = LDS_FRAG(w2s, (n * 16 + mrow) * K2 + k + khalf * 8);
      acc2[n] = __builtin_amdgcn_wmma_f32_16x16x32_f16(
          false, a2, false, b, (short)0, acc2[n], false, false);
    }
  }

  // LDS ops are in-order within a wave: overwriting h1 after its last read
  // is safe; fence before re-reading as h2.
  #pragma unroll
  for (int n = 0; n < 7; ++n) {
    int N = n * 16 + mrow;
    #pragma unroll
    for (int r = 0; r < 8; ++r) {
      int M = khalf * 8 + r;
      float v = acc2[n][r] + b2v[n];
      hbuf[hb + M * K2 + N] = (_Float16)(v > 0.f ? v : 0.f);
    }
  }
  asm volatile("s_wait_dscnt 0x0" ::: "memory");  // h2 visible within wave

  // ---------------- layer 3: [16x128] x [128x16] ----------------
  v8f acc3 = zacc;
  for (int k = 0; k < K2; k += 32) {
    v16h a3 = LDS_FRAG(hbuf, hb + mrow * K2 + k + khalf * 8);
    v16h b  = LDS_FRAG(w3s, mrow * K2 + k + khalf * 8);
    acc3 = __builtin_amdgcn_wmma_f32_16x16x32_f16(
        false, a3, false, b, (short)0, acc3, false, false);
  }

  // Write-once result: non-temporal stores keep L2 for weights/activations.
  if (mrow < 10) {
    #pragma unroll
    for (int r = 0; r < 8; ++r) {
      int M = khalf * 8 + r;
      __builtin_nontemporal_store(acc3[r] + b3v,
                                  out + (size_t)(m0 + M) * 10 + mrow);
    }
  }
}

// ---------------------------------------------------------------------------
extern "C" void kernel_launch(void* const* d_in, const int* in_sizes, int n_in,
                              void* d_out, int out_size, void* d_ws, size_t ws_size,
                              hipStream_t stream) {
  const float* x     = (const float*)d_in[0];
  const float* wconv = (const float*)d_in[1];
  const float* W1    = (const float*)d_in[2];
  const float* b1    = (const float*)d_in[3];
  const float* W2    = (const float*)d_in[4];
  const float* b2    = (const float*)d_in[5];
  const float* W3    = (const float*)d_in[6];
  const float* b3    = (const float*)d_in[7];
  float* out = (float*)d_out;

  char* ws = (char*)d_ws;
  _Float16* xh  = (_Float16*)(ws + WS_XH);
  _Float16* w1s = (_Float16*)(ws + WS_W1S);
  _Float16* w2t = (_Float16*)(ws + WS_W2T);
  _Float16* w3t = (_Float16*)(ws + WS_W3T);

  // 112*800 + 112*128 + 16*128 = 105,984 elements -> 414 blocks of 256
  prep_weights<<<414, 256, 0, stream>>>(wconv, W1, W2, W3, w1s, w2t, w3t);
  // 32768*100 chunks of 8 -> 12,800 blocks of 256
  cast_x<<<12800, 256, 0, stream>>>(x, xh);
  // 2048 strips of 16 rows, 8 waves/block -> 256 blocks of 256
  mlp_fused<<<256, 256, 0, stream>>>(xh, w1s, w2t, w3t, b1, b2, b3, out);
}
